// SupplyChainGNN_62191126446558
// MI455X (gfx1250) — matrix operand order repaired and checked by reading, not compiled, not run on hardware
//
#include <hip/hip_runtime.h>

// ---------------------------------------------------------------------------
// SupplyChainGNN for MI455X (gfx1250):
//   bf16 WMMA GEMMs + TDM (tensor_load_to_lds) tile staging + L2 scatter
// ---------------------------------------------------------------------------

typedef __bf16 v16bf __attribute__((ext_vector_type(16)));
typedef float  v8f   __attribute__((ext_vector_type(8)));
typedef unsigned int u32x4 __attribute__((ext_vector_type(4)));
typedef int          i32x4 __attribute__((ext_vector_type(4)));
typedef int          i32x8 __attribute__((ext_vector_type(8)));

#define N_NODES 100000
#define N_EDGES 1200000
#define IN_CH   128
#define HID     64

// TDM availability / arity detection (compile-safe on either toolchain)
#if defined(__has_builtin)
#  if __has_builtin(__builtin_amdgcn_tensor_load_to_lds) && \
      __has_builtin(__builtin_amdgcn_s_wait_tensorcnt)
#    define GNN_USE_TDM 1
#  endif
#endif
#ifndef GNN_USE_TDM
#  define GNN_USE_TDM 0
#endif
#if defined(__has_include)
#  if __has_include(<hip/amd_detail/amd_gfx1250_TDM.h>)
#    define GNN_TDM_6ARG 1
#  else
#    define GNN_TDM_6ARG 0
#  endif
#else
#  define GNN_TDM_6ARG 0
#endif

// Hardware f32->bf16 (fptrunc -> v_cvt_pk_bf16_f32, RNE on gfx1250)
__device__ __forceinline__ __bf16 f32_to_bf16_t(float f) {
  return (__bf16)f;
}
__device__ __forceinline__ unsigned short f32_to_bf16(float f) {
  return __builtin_bit_cast(unsigned short, (__bf16)f);
}

// ---------------------------------------------------------------------------
// Generic zero fill
// ---------------------------------------------------------------------------
__global__ void zero_f32(float* __restrict__ p, long long n) {
  for (long long i = (long long)blockIdx.x * blockDim.x + threadIdx.x;
       i < n; i += (long long)gridDim.x * blockDim.x)
    p[i] = 0.0f;
}

// ---------------------------------------------------------------------------
// Degree / normalization
// ---------------------------------------------------------------------------
__global__ void count_deg(const long long* __restrict__ ei, float* __restrict__ deg) {
  int e = blockIdx.x * blockDim.x + threadIdx.x;
  if (e < N_EDGES) atomicAdd(&deg[(int)ei[N_EDGES + e]], 1.0f);
}

__global__ void compute_dinv(const float* __restrict__ deg, float* __restrict__ dinv) {
  int i = blockIdx.x * blockDim.x + threadIdx.x;
  if (i < N_NODES) dinv[i] = rsqrtf(deg[i] + 1.0f);   // +1 = self loop
}

// ---------------------------------------------------------------------------
// GEMM: Y[N,64] = X[N,K] @ W[K,64]   (bf16 WMMA, f32 accumulate, no bias)
// Block = 128 threads = 4 waves; each wave computes a 16x64 strip.
// X tile staged global->LDS by the Tensor Data Mover (TENSORcnt), with the
// descriptor's pad feature inserting 1 DWORD per K DWORDs (row stride K+1
// floats -> consecutive rows land in adjacent LDS banks for A-frag reads).
// ---------------------------------------------------------------------------
template <int K>
__global__ void __launch_bounds__(128)
gemm_bf16_wmma(const float* __restrict__ X, const float* __restrict__ W,
               float* __restrict__ Y, int n_rows) {
  __shared__ float          XsF[64][K + 1];   // matches TDM pad layout
  __shared__ unsigned short Ws[K][HID + 2];   // stride 132 B (33 banks)

  const int tid  = threadIdx.x;
  const int row0 = blockIdx.x * 64;

  // Stage W (K x 64) as bf16 via normal loads (small, reused by all waves)
  for (int idx = tid; idx < K * HID; idx += 128) {
    int r = idx >> 6, c = idx & 63;
    Ws[r][c] = f32_to_bf16(W[idx]);
  }

#if GNN_USE_TDM
  // One TDM descriptor per block: 2-D tile, K x 64 f32, zero-fill OOB rows.
  if (tid < 32) {   // wave 0 only (whole wave active -> real branch)
    const unsigned lds_addr =
        (unsigned)(unsigned long long)(void*)&XsF[0][0];
    const unsigned long long ga =
        (unsigned long long)(const void*)(X + (size_t)row0 * K);
    const unsigned rem = (unsigned)(n_rows - row0);          // rows available
    constexpr unsigned PAD_IV = (K == 128) ? 6u : 5u;        // 2^(p+1) DWORDs = K

    u32x4 g0;
    g0.x = 0x1u;                                   // count=1 (valid), user mode
    g0.y = lds_addr;                               // lds_addr [63:32]
    g0.z = (unsigned)ga;                           // global_addr [95:64]
    g0.w = (unsigned)(ga >> 32) | (2u << 30);      // addr[56:32] | type=2

    i32x8 g1;
    g1[0] = (int)((2u << 16) | (1u << 20) | (PAD_IV << 22)); // 4B elems, pad 1 DW / K DW
    g1[1] = (int)((unsigned)K << 16);              // tensor_dim0[15:0] = K
    g1[2] = (int)((rem & 0xFFFFu) << 16);          // dim0 hi=0 | tensor_dim1 lo
    g1[3] = (int)((rem >> 16) | ((unsigned)K << 16)); // dim1 hi | tile_dim0 = K
    g1[4] = 64;                                    // tile_dim1=64, tile_dim2=0
    g1[5] = K;                                     // tensor_dim0_stride lo = K
    g1[6] = 0;                                     // stride hi | dim1_stride lo
    g1[7] = 0;

    i32x4 z4 = {0, 0, 0, 0};
#if GNN_TDM_6ARG
    i32x8 z8 = {0, 0, 0, 0, 0, 0, 0, 0};
    __builtin_amdgcn_tensor_load_to_lds(g0, g1, z4, z4, z8, 0);
#else
    __builtin_amdgcn_tensor_load_to_lds(g0, g1, z4, z4, 0);
#endif
    __builtin_amdgcn_s_wait_tensorcnt(0);
  }
#else
  // Fallback: VGPR staging with identical padded layout
  for (int idx = tid; idx < 64 * K; idx += 128) {
    int r = idx / K, c = idx % K;
    int g = row0 + r;
    XsF[r][c] = (g < n_rows) ? X[(size_t)g * K + c] : 0.0f;
  }
#endif
  __syncthreads();

  const int lane = tid & 31;
  const int wave = tid >> 5;
  const int m    = lane & 15;     // A row within tile
  const int kh   = lane >> 4;     // lane half selects K sub-group / C row-half
  const int wrow = wave * 16;

  const v8f vzero = {0.f, 0.f, 0.f, 0.f, 0.f, 0.f, 0.f, 0.f};
  v8f acc[4];
#pragma unroll
  for (int t = 0; t < 4; ++t) acc[t] = vzero;

#pragma unroll
  for (int kt = 0; kt < K / 32; ++kt) {
    const int kbase = kt * 32;

    // A fragment: 16x32 bf16 per ISA layout (hardware cvt f32->bf16 in regs).
    // VGPR v<4: K = 2v + 8*kh + off ; v>=4: K = 16 + 2(v-4) + 8*kh + off
    v16bf afrag;
#pragma unroll
    for (int j = 0; j < 16; ++j) {
      int v = j >> 1, off = j & 1;
      int base = (v < 4) ? (v * 2) : (16 + (v - 4) * 2);
      int k = kbase + base + kh * 8 + off;
      afrag[j] = f32_to_bf16_t(XsF[wrow + m][k]);
    }

    // B fragment: 32x16 bf16; lane holds column N = lane&15, 16 halfs = K (same order as A).
    const int krow = kbase + kh * 8;
#pragma unroll
    for (int nt = 0; nt < 4; ++nt) {
      v16bf bfrag;
#pragma unroll
      for (int j = 0; j < 16; ++j)
        bfrag[j] = __builtin_bit_cast(__bf16, Ws[krow + (j & 7) + ((j >> 3) << 4)][nt * 16 + m]);
      acc[nt] = __builtin_amdgcn_wmma_f32_16x16x32_bf16(
          false, afrag, false, bfrag, (short)0, acc[nt], false, false);
    }
  }

  // C/D layout: VGPR r, lane l -> row = r + 8*(l>>4), col = l&15
#pragma unroll
  for (int nt = 0; nt < 4; ++nt) {
#pragma unroll
    for (int r = 0; r < 8; ++r) {
      int g = row0 + wrow + r + kh * 8;
      if (g < n_rows)
        Y[(size_t)g * HID + nt * 16 + (lane & 15)] = acc[nt][r];
    }
  }
}

// ---------------------------------------------------------------------------
// Edge scatter: agg[dst,c] += h[src,c] * dinv[src]*dinv[dst]
// One thread per (edge, channel); gathers hit L2 (h = 25.6MB << 192MB L2).
// ---------------------------------------------------------------------------
__global__ void __launch_bounds__(256)
scatter_edges(const long long* __restrict__ ei, const float* __restrict__ h,
              const float* __restrict__ dinv, float* __restrict__ agg) {
  const long long total = (long long)N_EDGES * HID;
  for (long long i = (long long)blockIdx.x * blockDim.x + threadIdx.x;
       i < total; i += (long long)gridDim.x * blockDim.x) {
    int e = (int)(i >> 6);
    int c = (int)(i & 63);
    int s = (int)ei[e];
    int d = (int)ei[N_EDGES + e];
    float w = dinv[s] * dinv[d];
    atomicAdd(&agg[(size_t)d * HID + c], h[(size_t)s * HID + c] * w);
  }
}

// ---------------------------------------------------------------------------
// Finalize GCN output (self-loop + bias) and accumulate BN column statistics
// via LDS partial sums (ds_add_f32) -> 2*64 global atomics per block.
// ---------------------------------------------------------------------------
__global__ void __launch_bounds__(256)
finalize_stats(float* __restrict__ z, const float* __restrict__ h,
               const float* __restrict__ dinv, const float* __restrict__ bias,
               float* __restrict__ stats) {
  __shared__ float s_sum[HID];
  __shared__ float s_sq[HID];
  if (threadIdx.x < HID) { s_sum[threadIdx.x] = 0.f; s_sq[threadIdx.x] = 0.f; }
  __syncthreads();

  const long long total = (long long)N_NODES * HID;
  for (long long i = (long long)blockIdx.x * blockDim.x + threadIdx.x;
       i < total; i += (long long)gridDim.x * blockDim.x) {
    int node = (int)(i >> 6), c = (int)(i & 63);
    float di = dinv[node];
    float v = z[i] + h[i] * di * di + bias[c];
    z[i] = v;
    atomicAdd(&s_sum[c], v);
    atomicAdd(&s_sq[c], v * v);
  }
  __syncthreads();
  if (threadIdx.x < HID) {
    atomicAdd(&stats[threadIdx.x], s_sum[threadIdx.x]);
    atomicAdd(&stats[HID + threadIdx.x], s_sq[threadIdx.x]);
  }
}

// BN params: a = g*rsqrt(var+eps), b = be - mu*a
__global__ void bn_params(const float* __restrict__ stats, const float* __restrict__ g,
                          const float* __restrict__ be, float* __restrict__ ab) {
  int c = threadIdx.x;
  if (c < HID) {
    const float inv_n = 1.0f / (float)N_NODES;
    float mu  = stats[c] * inv_n;
    float var = stats[HID + c] * inv_n - mu * mu;
    float rs  = rsqrtf(var + 1e-5f);
    float a   = rs * g[c];
    ab[c]       = a;
    ab[HID + c] = be[c] - mu * a;
  }
}

__global__ void __launch_bounds__(256)
bn_relu_apply(float* __restrict__ z, const float* __restrict__ ab) {
  const long long total = (long long)N_NODES * HID;
  for (long long i = (long long)blockIdx.x * blockDim.x + threadIdx.x;
       i < total; i += (long long)gridDim.x * blockDim.x) {
    int c = (int)(i & 63);
    z[i] = fmaxf(fmaf(z[i], ab[c], ab[HID + c]), 0.0f);
  }
}

// ---------------------------------------------------------------------------
// Fused MLP head: 64 -> 32 -> 16 -> 1 (+sigmoid); weights staged in LDS.
// ---------------------------------------------------------------------------
__global__ void __launch_bounds__(256)
mlp_head(const float* __restrict__ h,
         const float* __restrict__ Wm1, const float* __restrict__ bm1,
         const float* __restrict__ Wm2, const float* __restrict__ bm2,
         const float* __restrict__ Wm3, const float* __restrict__ bm3,
         float* __restrict__ out) {
  __shared__ float w1[64 * 32];
  __shared__ float w2[32 * 16];
  __shared__ float w3[16], b1s[32], b2s[16], b3s;

  for (int i = threadIdx.x; i < 64 * 32; i += 256) w1[i] = Wm1[i];
  for (int i = threadIdx.x; i < 32 * 16; i += 256) w2[i] = Wm2[i];
  if (threadIdx.x < 16) w3[threadIdx.x]  = Wm3[threadIdx.x];
  if (threadIdx.x < 32) b1s[threadIdx.x] = bm1[threadIdx.x];
  if (threadIdx.x < 16) b2s[threadIdx.x] = bm2[threadIdx.x];
  if (threadIdx.x == 0) b3s = bm3[0];
  __syncthreads();

  for (int node = blockIdx.x * blockDim.x + threadIdx.x;
       node < N_NODES; node += gridDim.x * blockDim.x) {
    const float* hp = h + (size_t)node * HID;
    float hv[64];
#pragma unroll
    for (int c = 0; c < 64; ++c) hv[c] = hp[c];

    float m1[32];
#pragma unroll
    for (int o = 0; o < 32; ++o) {
      float acc = b1s[o];
#pragma unroll
      for (int c = 0; c < 64; ++c) acc = fmaf(hv[c], w1[c * 32 + o], acc);
      m1[o] = fmaxf(acc, 0.f);
    }
    float m2[16];
#pragma unroll
    for (int o = 0; o < 16; ++o) {
      float acc = b2s[o];
#pragma unroll
      for (int c = 0; c < 32; ++c) acc = fmaf(m1[c], w2[c * 16 + o], acc);
      m2[o] = fmaxf(acc, 0.f);
    }
    float acc = b3s;
#pragma unroll
    for (int c = 0; c < 16; ++c) acc = fmaf(m2[c], w3[c], acc);
    out[node] = 1.0f / (1.0f + __expf(-acc));
  }
}

// ---------------------------------------------------------------------------
// Host-side orchestration (graph-capture safe: stream-only, no sync/allocs)
// ---------------------------------------------------------------------------
extern "C" void kernel_launch(void* const* d_in, const int* in_sizes, int n_in,
                              void* d_out, int out_size, void* d_ws, size_t ws_size,
                              hipStream_t stream) {
  const float*     x   = (const float*)d_in[0];
  const long long* ei  = (const long long*)d_in[1];
  const float*     W1  = (const float*)d_in[2];
  const float*     b1  = (const float*)d_in[3];
  const float*     g1  = (const float*)d_in[4];
  const float*     be1 = (const float*)d_in[5];
  const float*     W2  = (const float*)d_in[6];
  const float*     b2  = (const float*)d_in[7];
  const float*     g2  = (const float*)d_in[8];
  const float*     be2 = (const float*)d_in[9];
  const float*     Wm1 = (const float*)d_in[10];
  const float*     bm1 = (const float*)d_in[11];
  const float*     Wm2 = (const float*)d_in[12];
  const float*     bm2 = (const float*)d_in[13];
  const float*     Wm3 = (const float*)d_in[14];
  const float*     bm3 = (const float*)d_in[15];
  float*           out = (float*)d_out;

  // Workspace layout (floats): deg | dinv | buf0 | buf1 | stats | bn_ab
  float* ws    = (float*)d_ws;
  float* deg   = ws;
  float* dinv  = deg  + N_NODES;
  float* buf0  = dinv + N_NODES;                      // GEMM output h
  float* buf1  = buf0 + (size_t)N_NODES * HID;        // aggregation / activations
  float* stats = buf1 + (size_t)N_NODES * HID;        // 2*HID
  float* bnab  = stats + 2 * HID;                     // 2*HID

  const long long featN = (long long)N_NODES * HID;
  const int gemm_grid = (N_NODES + 63) / 64;

  // --- degree normalization ---
  zero_f32<<<512, 256, 0, stream>>>(deg, N_NODES);
  count_deg<<<(N_EDGES + 255) / 256, 256, 0, stream>>>(ei, deg);
  compute_dinv<<<(N_NODES + 255) / 256, 256, 0, stream>>>(deg, dinv);

  // --- GCN layer 1 ---
  gemm_bf16_wmma<IN_CH><<<gemm_grid, 128, 0, stream>>>(x, W1, buf0, N_NODES);
  zero_f32<<<2048, 256, 0, stream>>>(buf1, featN);
  scatter_edges<<<4096, 256, 0, stream>>>(ei, buf0, dinv, buf1);
  zero_f32<<<1, 256, 0, stream>>>(stats, 2 * HID);
  finalize_stats<<<2048, 256, 0, stream>>>(buf1, buf0, dinv, b1, stats);
  bn_params<<<1, 64, 0, stream>>>(stats, g1, be1, bnab);
  bn_relu_apply<<<2048, 256, 0, stream>>>(buf1, bnab);

  // --- GCN layer 2 ---
  gemm_bf16_wmma<HID><<<gemm_grid, 128, 0, stream>>>(buf1, W2, buf0, N_NODES);
  zero_f32<<<2048, 256, 0, stream>>>(buf1, featN);
  scatter_edges<<<4096, 256, 0, stream>>>(ei, buf0, dinv, buf1);
  zero_f32<<<1, 256, 0, stream>>>(stats, 2 * HID);
  finalize_stats<<<2048, 256, 0, stream>>>(buf1, buf0, dinv, b2, stats);
  bn_params<<<1, 64, 0, stream>>>(stats, g2, be2, bnab);
  bn_relu_apply<<<2048, 256, 0, stream>>>(buf1, bnab);

  // --- MLP head ---
  mlp_head<<<1024, 256, 0, stream>>>(buf1, Wm1, bm1, Wm2, bm2, Wm3, bm3, out);
}
